// AttentionBlock_14250701488245
// MI455X (gfx1250) — matrix operand order repaired and checked by reading, not compiled, not run on hardware
//
#include <hip/hip_runtime.h>
#include <hip/hip_fp16.h>
#include <hip/hip_bf16.h>

typedef __attribute__((ext_vector_type(16))) _Float16 v16h;
typedef __attribute__((ext_vector_type(8)))  float    v8f;
typedef __attribute__((ext_vector_type(4)))  unsigned int v4u;

#define TOKENS 8192   // B*H*W
#define NPB    4096   // tokens per batch
#define CDIM   512
#define NBATCH 2
#define GROUPS 32

// ---------------- CDNA5 (gfx1250) primitives ----------------

__device__ __forceinline__ v8f wmma_f16(v16h a, v16h b, v8f c) {
  // D = A(16x32 f16) * B(32x16 f16) + C(16x16 f32)
  return __builtin_amdgcn_wmma_f32_16x16x32_f16(
      false, a, false, b, (short)0, c, false, false);
}

// Async DMA: one 16B chunk, global -> LDS (tracked by ASYNCcnt).
// lds_off is the byte offset from the wave's LDS base (dynamic LDS starts at 0
// when the kernel has no static LDS).
__device__ __forceinline__ void async_ld_b128(unsigned lds_off, const void* g) {
  asm volatile("global_load_async_to_lds_b128 %0, %1, off"
               :: "v"(lds_off), "v"(g)
               : "memory");
}
__device__ __forceinline__ void wait_async0() {
  asm volatile("s_wait_asynccnt 0x0" ::: "memory");
}

// A fragment (16x32 f16): lane L holds M=L%16, K = {off..off+7, off+16..off+23},
// off = (L<16 ? 0 : 8).  Two 16B LDS loads.
__device__ __forceinline__ v16h frag_a(const _Float16* row, int off) {
  union { v4u u[2]; v16h h; } t;
  t.u[0] = *(const v4u*)(row + off);
  t.u[1] = *(const v4u*)(row + off + 16);
  return t.h;
}

// B fragment (32x16 f16): lane L holds N=L%16, K = {koff/2..koff/2+7, koff/2+16..koff/2+23},
// koff = (L<16 ? 0 : 16).  Two 16B runs.
__device__ __forceinline__ v16h frag_b(const _Float16* row, int koff) {
  union { v4u u[2]; v16h h; } t;
  t.u[0] = *(const v4u*)(row + koff / 2);
  t.u[1] = *(const v4u*)(row + koff / 2 + 16);
  return t.h;
}

// ---------------- 1) weight convert: f32 [cin][cout] -> f16 [w][cout][cin] ----

__global__ void __launch_bounds__(256)
wcvt_kernel(const float* __restrict__ wq, const float* __restrict__ wk,
            const float* __restrict__ wv, const float* __restrict__ wp,
            _Float16* __restrict__ wt) {
  int idx  = blockIdx.x * 256 + threadIdx.x;        // 4*512*512 total
  int wi   = idx >> 18;
  int rem  = idx & ((1 << 18) - 1);
  int cout = rem >> 9;
  int cin  = rem & 511;
  const float* w = (wi == 0) ? wq : (wi == 1) ? wk : (wi == 2) ? wv : wp;
  wt[idx] = (_Float16)w[cin * CDIM + cout];
}

// ---------------- 2) groupnorm stats ----------------------------------------

__global__ void __launch_bounds__(256)
gn_stats_kernel(const float* __restrict__ x, float* __restrict__ stats) {
  int bg = blockIdx.x;           // 0..63
  int b  = bg >> 5, g = bg & 31;
  const float* base = x + (size_t)b * NPB * CDIM + g * 16;
  float sum = 0.f, sq = 0.f;
  for (int p = threadIdx.x; p < NPB; p += 256) {
    const float4* v = (const float4*)(base + (size_t)p * CDIM);
#pragma unroll
    for (int i = 0; i < 4; ++i) {
      float4 f = v[i];
      sum += f.x + f.y + f.z + f.w;
      sq  += f.x * f.x + f.y * f.y + f.z * f.z + f.w * f.w;
    }
  }
#pragma unroll
  for (int m = 1; m < 32; m <<= 1) {
    sum += __shfl_xor(sum, m, 32);
    sq  += __shfl_xor(sq,  m, 32);
  }
  __shared__ float ssum[8], ssq[8];
  if ((threadIdx.x & 31) == 0) { ssum[threadIdx.x >> 5] = sum; ssq[threadIdx.x >> 5] = sq; }
  __syncthreads();
  if (threadIdx.x == 0) {
    float S = 0.f, Q = 0.f;
#pragma unroll
    for (int i = 0; i < 8; ++i) { S += ssum[i]; Q += ssq[i]; }
    float mean = S * (1.f / 65536.f);
    float var  = Q * (1.f / 65536.f) - mean * mean;
    stats[bg]      = mean;
    stats[64 + bg] = rsqrtf(var + 1e-5f);
  }
}

// ---------------- 3) groupnorm apply -> xn f16 ------------------------------

__global__ void __launch_bounds__(256)
gn_apply_kernel(const float* __restrict__ x, const float* __restrict__ gamma,
                const float* __restrict__ beta, const float* __restrict__ stats,
                _Float16* __restrict__ xnh) {
  size_t i4 = ((size_t)blockIdx.x * 256 + threadIdx.x) * 4;
  int c = (int)(i4 & (CDIM - 1));
  size_t m = i4 >> 9;
  int b = (int)(m >> 12);
  int bg = b * GROUPS + (c >> 4);
  float mean = stats[bg], rstd = stats[64 + bg];
  float4 xv = *(const float4*)(x + i4);
  float4 gv = *(const float4*)(gamma + c);
  float4 bv = *(const float4*)(beta + c);
  union { _Float16 h[4]; unsigned long long u; } t;
  t.h[0] = (_Float16)((xv.x - mean) * rstd * gv.x + bv.x);
  t.h[1] = (_Float16)((xv.y - mean) * rstd * gv.y + bv.y);
  t.h[2] = (_Float16)((xv.z - mean) * rstd * gv.z + bv.z);
  t.h[3] = (_Float16)((xv.w - mean) * rstd * gv.w + bv.w);
  *(unsigned long long*)(xnh + i4) = t.u;
}

// ---------------- 4) QKV GEMM, async double-buffered LDS --------------------
// dynamic LDS: X0 @0, X1 @4K, W0 @8K, W1 @12K (16 KB total)

__global__ void __launch_bounds__(256)
gemm_qkv_kernel(const _Float16* __restrict__ xnh, const _Float16* __restrict__ wt,
                const float* __restrict__ bq, const float* __restrict__ bk,
                const float* __restrict__ bv,
                _Float16* __restrict__ qh, _Float16* __restrict__ kh,
                _Float16* __restrict__ vth) {
  extern __shared__ char gsm[];
  const int mode = blockIdx.z;
  const int m0 = blockIdx.x * 64;
  const int n0 = blockIdx.y * 64;
  const _Float16* w = wt + (size_t)mode * CDIM * CDIM;
  const float* bias = (mode == 0) ? bq : (mode == 1) ? bk : bv;
  const int tid = threadIdx.x, wave = tid >> 5, lane = tid & 31;
  const int hh = lane >> 4, l16 = lane & 15;
  const int mt = wave & 3, ntA = (wave >> 2) * 2;
  const int lrow = tid >> 2, lg = tid & 3;
  const unsigned loff = (unsigned)(lrow * 32 + lg * 8) * 2u;  // byte off in tile

  auto issue = [&](int kc, int buf) {
    async_ld_b128(buf * 4096u + loff,
                  xnh + (size_t)(m0 + lrow) * CDIM + kc + lg * 8);
    async_ld_b128(8192u + buf * 4096u + loff,
                  w + (size_t)(n0 + lrow) * CDIM + kc + lg * 8);
  };

  v8f acc0 = {0.f, 0.f, 0.f, 0.f, 0.f, 0.f, 0.f, 0.f};
  v8f acc1 = acc0;

  issue(0, 0);
  for (int kc = 0; kc < CDIM; kc += 32) {
    const int buf = (kc >> 5) & 1;
    wait_async0();       // current buffer's DMA complete (this wave)
    __syncthreads();     // ... and everyone else's
    if (kc + 32 < CDIM) issue(kc + 32, buf ^ 1);
    if (kc + 64 < CDIM)  // warm L2 for the tile after next
      __builtin_prefetch(xnh + (size_t)(m0 + lrow) * CDIM + kc + 64, 0, 1);
    const _Float16* Xb = (const _Float16*)(gsm + buf * 4096);
    const _Float16* Wb = (const _Float16*)(gsm + 8192 + buf * 4096);
    v16h a  = frag_a(Xb + (mt * 16 + l16) * 32, hh ? 8 : 0);
    v16h b0 = frag_b(Wb + ((ntA)     * 16 + l16) * 32, hh ? 16 : 0);
    v16h b1 = frag_b(Wb + ((ntA + 1) * 16 + l16) * 32, hh ? 16 : 0);
    acc0 = wmma_f16(a, b0, acc0);
    acc1 = wmma_f16(a, b1, acc1);
  }

  const float scale = 0.04419417382415922f;  // 512^-0.5 folded into q
#pragma unroll
  for (int t = 0; t < 2; ++t) {
    v8f acc = t ? acc1 : acc0;
    int n = n0 + (ntA + t) * 16 + l16;
    float bn = bias[n];
    int mrow0 = m0 + mt * 16 + hh * 8;
    if (mode == 0) {
#pragma unroll
      for (int r = 0; r < 8; ++r)
        qh[(size_t)(mrow0 + r) * CDIM + n] = (_Float16)((acc[r] + bn) * scale);
    } else if (mode == 1) {
#pragma unroll
      for (int r = 0; r < 8; ++r)
        kh[(size_t)(mrow0 + r) * CDIM + n] = (_Float16)(acc[r] + bn);
    } else {
      alignas(16) _Float16 tmp[8];
#pragma unroll
      for (int r = 0; r < 8; ++r) tmp[r] = (_Float16)(acc[r] + bn);
      int b = mrow0 >> 12, p0 = mrow0 & (NPB - 1);
      *(v4u*)(vth + ((size_t)b * CDIM + n) * NPB + p0) = *(const v4u*)tmp;
    }
  }
}

// ---------------- 5) flash attention, async double-buffered K/V -------------
// dynamic LDS layout (bytes):

#define OFF_QS 0u                         // 64 x 512 f16   (64 KB)
#define OFF_K0 65536u                     // 32 x 512 f16   (32 KB)
#define OFF_K1 (OFF_K0 + 32768u)
#define OFF_V0 (OFF_K1 + 32768u)          // 512 x 32 f16   (32 KB)  V^T slice
#define OFF_V1 (OFF_V0 + 32768u)
#define OFF_PS (OFF_V1 + 32768u)          // 64 x 32 f16    (4 KB)
#define OFF_ST (OFF_PS + 4096u)           // 512 floats     (2 KB)
#define SMEM_FLASH (OFF_ST + 2048u)       // 202752 B  (< 320 KB/WGP)

__global__ void __launch_bounds__(256)
flash_attn_kernel(const _Float16* __restrict__ qh, const _Float16* __restrict__ kh,
                  const _Float16* __restrict__ vtp, _Float16* __restrict__ aoh) {
  extern __shared__ char smem[];
  _Float16* Qs = (_Float16*)(smem + OFF_QS);
  _Float16* Ps = (_Float16*)(smem + OFF_PS);
  float* s_m  = (float*)(smem + OFF_ST);   // running row max   [64]
  float* s_l  = s_m + 64;                  // running row sum   [64]
  float* s_nm = s_l + 64;                  // new max           [64]
  float* s_al = s_nm + 64;                 // alpha             [64]
  float* s_pm = s_al + 64;                 // partial max    [2][64]
  float* s_ps = s_pm + 128;                // partial sum    [2][64]

  const int b  = blockIdx.y;
  const int q0 = blockIdx.x * 64;
  const int tid = threadIdx.x, wave = tid >> 5, lane = tid & 31;
  const int hh = lane >> 4, l16 = lane & 15;
  const int mt = wave & 3, nh = wave >> 2;
  const size_t tok0 = (size_t)b * NPB + q0;

  auto issue_kv = [&](int j, int buf) {
    const _Float16* ksrc = kh + ((size_t)b * NPB + j * 32) * CDIM;
    const unsigned kbase = buf ? OFF_K1 : OFF_K0;
    for (int i = tid; i < 2048; i += 256)             // 32 KB contiguous
      async_ld_b128(kbase + (unsigned)i * 16u, ksrc + (size_t)i * 8);
    const _Float16* vsrc = vtp + (size_t)b * CDIM * NPB + j * 32;
    const unsigned vbase = buf ? OFF_V1 : OFF_V0;
    for (int i = tid; i < 2048; i += 256) {           // 512 rows x 64 B
      int c = i >> 2, g = i & 3;
      async_ld_b128(vbase + (unsigned)(c * 32 + g * 8) * 2u,
                    vsrc + (size_t)c * NPB + g * 8);
    }
  };

  // Q tile (64 x 512, contiguous tokens) via async DMA; K/V tile 0 behind it.
  {
    const _Float16* qsrc = qh + tok0 * CDIM;
    for (int i = tid; i < 4096; i += 256)
      async_ld_b128(OFF_QS + (unsigned)i * 16u, qsrc + (size_t)i * 8);
  }
  issue_kv(0, 0);
  if (tid < 64) { s_m[tid] = -1e30f; s_l[tid] = 0.f; }

  v8f o[16];
#pragma unroll
  for (int t = 0; t < 16; ++t) o[t] = (v8f){0.f, 0.f, 0.f, 0.f, 0.f, 0.f, 0.f, 0.f};

  for (int j = 0; j < NPB / 32; ++j) {
    const int buf = j & 1;
    wait_async0();       // Q (j==0) + current K/V buffer landed (this wave)
    __syncthreads();     // (A) global: DMAs landed, prev iter done with bufs
    if (j + 1 < NPB / 32) issue_kv(j + 1, buf ^ 1);

    const _Float16* Kb = (const _Float16*)(smem + (buf ? OFF_K1 : OFF_K0));
    const _Float16* Vb = (const _Float16*)(smem + (buf ? OFF_V1 : OFF_V0));

    // S tile 16x16: queries mt*16.., keys nh*16..  (16 WMMAs over d=512)
    v8f s = {0.f, 0.f, 0.f, 0.f, 0.f, 0.f, 0.f, 0.f};
#pragma unroll
    for (int kc = 0; kc < CDIM; kc += 32) {
      v16h a  = frag_a(Qs + (mt * 16 + l16) * 512 + kc, hh ? 8 : 0);
      v16h bb = frag_b(Kb + (nh * 16 + l16) * 512 + kc, hh ? 16 : 0);
      s = wmma_f16(a, bb, s);
    }

    // per-row max within the 16-lane half (row = r + 8*hh in C/D layout)
#pragma unroll
    for (int r = 0; r < 8; ++r) {
      float v = s[r];
      v = fmaxf(v, __shfl_xor(v, 1, 32));
      v = fmaxf(v, __shfl_xor(v, 2, 32));
      v = fmaxf(v, __shfl_xor(v, 4, 32));
      v = fmaxf(v, __shfl_xor(v, 8, 32));
      if (l16 == 0) s_pm[nh * 64 + mt * 16 + hh * 8 + r] = v;
    }
    __syncthreads();  // (B)
    if (tid < 64) {
      float mo = s_m[tid];
      float mn = fmaxf(mo, fmaxf(s_pm[tid], s_pm[64 + tid]));
      s_nm[tid] = mn;
      s_al[tid] = __expf(mo - mn);
      s_m[tid]  = mn;
    }
    __syncthreads();  // (C)

    // P = exp(S - newmax) -> Ps (f16) + partial row sums
#pragma unroll
    for (int r = 0; r < 8; ++r) {
      int row = mt * 16 + hh * 8 + r;
      float p = __expf(s[r] - s_nm[row]);
      Ps[row * 32 + nh * 16 + l16] = (_Float16)p;
      float v = p;
      v += __shfl_xor(v, 1, 32);
      v += __shfl_xor(v, 2, 32);
      v += __shfl_xor(v, 4, 32);
      v += __shfl_xor(v, 8, 32);
      if (l16 == 0) s_ps[nh * 64 + row] = v;
    }
    // rescale O accumulators by alpha
#pragma unroll
    for (int tmt = 0; tmt < 4; ++tmt) {
      float al[8];
#pragma unroll
      for (int r = 0; r < 8; ++r) al[r] = s_al[tmt * 16 + hh * 8 + r];
#pragma unroll
      for (int ct = 0; ct < 4; ++ct)
#pragma unroll
        for (int r = 0; r < 8; ++r) o[tmt * 4 + ct][r] *= al[r];
    }
    __syncthreads();  // (D) Ps + partial sums visible
    if (tid < 64) s_l[tid] = s_l[tid] * s_al[tid] + s_ps[tid] + s_ps[64 + tid];

    // O += P @ V  (wave's 64-wide d-slice, 16 WMMAs)
#pragma unroll
    for (int tmt = 0; tmt < 4; ++tmt) {
      v16h a = frag_a(Ps + (tmt * 16 + l16) * 32, hh ? 8 : 0);
#pragma unroll
      for (int ct = 0; ct < 4; ++ct) {
        v16h bb = frag_b(Vb + (wave * 64 + ct * 16 + l16) * 32, hh ? 16 : 0);
        o[tmt * 4 + ct] = wmma_f16(a, bb, o[tmt * 4 + ct]);
      }
    }
  }
  __syncthreads();

  // finalize: O / l -> f16 row-major
#pragma unroll
  for (int tmt = 0; tmt < 4; ++tmt) {
    float inv[8];
#pragma unroll
    for (int r = 0; r < 8; ++r) inv[r] = 1.f / s_l[tmt * 16 + hh * 8 + r];
#pragma unroll
    for (int ct = 0; ct < 4; ++ct) {
      int c = wave * 64 + ct * 16 + l16;
#pragma unroll
      for (int r = 0; r < 8; ++r) {
        int row = tmt * 16 + hh * 8 + r;
        aoh[(tok0 + row) * CDIM + c] = (_Float16)(o[tmt * 4 + ct][r] * inv[r]);
      }
    }
  }
}

// ---------------- 6) output projection + bias + residual --------------------

__global__ void __launch_bounds__(256)
gemm_proj_kernel(const _Float16* __restrict__ aoh, const _Float16* __restrict__ wpt,
                 const float* __restrict__ bp, const float* __restrict__ x,
                 float* __restrict__ out) {
  extern __shared__ char gsm[];
  const int m0 = blockIdx.x * 64;
  const int n0 = blockIdx.y * 64;
  const int tid = threadIdx.x, wave = tid >> 5, lane = tid & 31;
  const int hh = lane >> 4, l16 = lane & 15;
  const int mt = wave & 3, ntA = (wave >> 2) * 2;
  const int lrow = tid >> 2, lg = tid & 3;
  const unsigned loff = (unsigned)(lrow * 32 + lg * 8) * 2u;

  auto issue = [&](int kc, int buf) {
    async_ld_b128(buf * 4096u + loff,
                  aoh + (size_t)(m0 + lrow) * CDIM + kc + lg * 8);
    async_ld_b128(8192u + buf * 4096u + loff,
                  wpt + (size_t)(n0 + lrow) * CDIM + kc + lg * 8);
  };

  v8f acc0 = {0.f, 0.f, 0.f, 0.f, 0.f, 0.f, 0.f, 0.f};
  v8f acc1 = acc0;

  issue(0, 0);
  for (int kc = 0; kc < CDIM; kc += 32) {
    const int buf = (kc >> 5) & 1;
    wait_async0();
    __syncthreads();
    if (kc + 32 < CDIM) issue(kc + 32, buf ^ 1);
    const _Float16* Xb = (const _Float16*)(gsm + buf * 4096);
    const _Float16* Wb = (const _Float16*)(gsm + 8192 + buf * 4096);
    v16h a  = frag_a(Xb + (mt * 16 + l16) * 32, hh ? 8 : 0);
    v16h b0 = frag_b(Wb + ((ntA)     * 16 + l16) * 32, hh ? 16 : 0);
    v16h b1 = frag_b(Wb + ((ntA + 1) * 16 + l16) * 32, hh ? 16 : 0);
    acc0 = wmma_f16(a, b0, acc0);
    acc1 = wmma_f16(a, b1, acc1);
  }

#pragma unroll
  for (int t = 0; t < 2; ++t) {
    v8f acc = t ? acc1 : acc0;
    int n = n0 + (ntA + t) * 16 + l16;
    float bn = bp[n];
    int mrow0 = m0 + mt * 16 + hh * 8;
#pragma unroll
    for (int r = 0; r < 8; ++r) {
      size_t idx = (size_t)(mrow0 + r) * CDIM + n;
      out[idx] = acc[r] + bn + x[idx];
    }
  }
}

// ---------------- host launch ----------------

extern "C" void kernel_launch(void* const* d_in, const int* in_sizes, int n_in,
                              void* d_out, int out_size, void* d_ws, size_t ws_size,
                              hipStream_t stream) {
  (void)in_sizes; (void)n_in; (void)out_size; (void)ws_size;
  const float* x     = (const float*)d_in[0];
  const float* gamma = (const float*)d_in[1];
  const float* beta  = (const float*)d_in[2];
  const float* wq    = (const float*)d_in[3];
  const float* bq    = (const float*)d_in[4];
  const float* wk    = (const float*)d_in[5];
  const float* bk    = (const float*)d_in[6];
  const float* wv    = (const float*)d_in[7];
  const float* bv    = (const float*)d_in[8];
  const float* wp    = (const float*)d_in[9];
  const float* bp    = (const float*)d_in[10];
  float* out = (float*)d_out;

  char* ws = (char*)d_ws;
  const size_t SZ_H = (size_t)TOKENS * CDIM * sizeof(_Float16);  // 8 MB
  _Float16* xnh = (_Float16*)(ws);
  _Float16* qh  = (_Float16*)(ws + 1 * SZ_H);
  _Float16* kh  = (_Float16*)(ws + 2 * SZ_H);
  _Float16* vth = (_Float16*)(ws + 3 * SZ_H);   // [b][c][n] transposed V
  _Float16* aoh = (_Float16*)(ws + 4 * SZ_H);
  _Float16* wt4 = (_Float16*)(ws + 5 * SZ_H);   // 4 x [cout][cin] f16
  float* stats  = (float*)(ws + 5 * SZ_H + 4 * (size_t)CDIM * CDIM * sizeof(_Float16));

  wcvt_kernel<<<4 * CDIM * CDIM / 256, 256, 0, stream>>>(wq, wk, wv, wp, wt4);
  gn_stats_kernel<<<NBATCH * GROUPS, 256, 0, stream>>>(x, stats);
  gn_apply_kernel<<<TOKENS * CDIM / 1024, 256, 0, stream>>>(x, gamma, beta, stats, xnh);
  gemm_qkv_kernel<<<dim3(TOKENS / 64, CDIM / 64, 3), 256, 16384, stream>>>(
      xnh, wt4, bq, bk, bv, qh, kh, vth);
  flash_attn_kernel<<<dim3(NPB / 64, NBATCH), 256, SMEM_FLASH, stream>>>(
      qh, kh, vth, aoh);
  gemm_proj_kernel<<<dim3(TOKENS / 64, CDIM / 64), 256, 16384, stream>>>(
      aoh, wt4 + 3 * (size_t)CDIM * CDIM, bp, x, out);
}
